// MoonElecToNucGamma_39161511804980
// MI455X (gfx1250) — matrix operand order repaired and checked by reading, not compiled, not run on hardware
//
#include <hip/hip_runtime.h>

typedef __attribute__((ext_vector_type(16))) _Float16 v16h;
typedef __attribute__((ext_vector_type(8)))  _Float16 v8h;
typedef __attribute__((ext_vector_type(8)))  float    v8f;

#define N_NUC  256
#define N_NB   1024
#define N_ENV  8
#define F0     32
#define F1     16
#define FEAT   256
#define CUTOFF 5.0f

__global__ __launch_bounds__(256)
void moon_gamma_edge_kernel(const float* __restrict__ r_nb_ne,   // (256,1024,3)
                            const float* __restrict__ R,         // (256,3)
                            const float* __restrict__ ne_scales, // (256,8)
                            const float* __restrict__ ne_kernel, // (256,4,32)
                            const float* __restrict__ ne_bias,   // (256,32)
                            const float* __restrict__ env_weight,// (8,32)
                            const float* __restrict__ W_beta,    // (32,16)
                            const float* __restrict__ W_gamma,   // (16,256)
                            const float* __restrict__ W_edge,    // (4,256)
                            const float* __restrict__ b_edge,    // (256)
                            const float* __restrict__ z_n,       // (256,256)
                            float* __restrict__ gamma_out,       // (256,1024,256)
                            float* __restrict__ edge_out)        // (256,1024,256)
{
    // ---- LDS staging (~61 KB total) ----
    __shared__ __align__(16) _Float16 s_beta[N_NB * F1];          // 32 KB, beta in f16, row-major (e, k)
    __shared__ __align__(16) float    s_inp[N_NB * 4];            // 16 KB, edge input vec per electron
    __shared__ __align__(16) _Float16 s_wgT[(FEAT + 16) * F1];    // 8.5 KB, WgT[f][k] + 16 zero pad rows
    __shared__ __align__(16) float    s_bz[FEAT];                 // 1 KB  (b_edge + z_n)
    __shared__ __align__(16) float    s_wbeta[F0 * F1];           // 2 KB
    __shared__ __align__(16) float    s_kern[4 * F0];             // 512 B
    __shared__ __align__(16) float    s_envw[N_ENV * F0];         // 1 KB
    __shared__ float s_bias[F0];
    __shared__ float s_scales[N_ENV];

    const int tid = threadIdx.x;
    const int n   = blockIdx.x;

    // ---- Phase 0: cooperative load of per-nucleus + shared weights ----
    if (tid < 4 * F0)  s_kern[tid]   = ne_kernel[n * 4 * F0 + tid];
    if (tid < F0)      s_bias[tid]   = ne_bias[n * F0 + tid];
    if (tid < N_ENV)   s_scales[tid] = ne_scales[n * N_ENV + tid];
    s_envw[tid]  = env_weight[tid];                 // 8*32 == 256
    s_wbeta[tid]        = W_beta[tid];              // 32*16 == 512
    s_wbeta[tid + 256]  = W_beta[tid + 256];
    s_bz[tid] = b_edge[tid] + z_n[n * FEAT + tid];
    // transpose W_gamma (k-major) -> WgT[f][k] as f16
    for (int i = tid; i < F1 * FEAT; i += 256) {
        int f = i >> 4, k = i & 15;
        s_wgT[f * F1 + k] = (_Float16)W_gamma[k * FEAT + f];
    }
    s_wgT[FEAT * F1 + tid] = (_Float16)0.0f;        // zero the 16 pad rows (256 halfs)

    __syncthreads();

    const float Rx = R[n * 3 + 0], Ry = R[n * 3 + 1], Rz = R[n * 3 + 2];

    // ---- Phase 1: per-electron beta(16) and edge inputs; 4 electrons/thread ----
    float dd[4][4];        // [dist, dx, dy, dz]
    float evv[4][N_ENV];   // exp(-dist*scale_v)
    float bacc[4][F1];
    float cutv[4];

    #pragma unroll
    for (int q = 0; q < 4; ++q) {
        const int e = q * 256 + tid;
        const float* rp = r_nb_ne + ((size_t)n * N_NB + e) * 3;
        const float dx = rp[0] - Rx, dy = rp[1] - Ry, dz = rp[2] - Rz;
        const float dist = sqrtf(dx * dx + dy * dy + dz * dz);
        dd[q][0] = dist; dd[q][1] = dx; dd[q][2] = dy; dd[q][3] = dz;
        #pragma unroll
        for (int v = 0; v < N_ENV; ++v) evv[q][v] = __expf(-dist * s_scales[v]);
        #pragma unroll
        for (int j = 0; j < F1; ++j) bacc[q][j] = 0.0f;
        const float x  = dist * (1.0f / CUTOFF);
        const float om = 1.0f - x;
        cutv[q] = (x < 1.0f) ? om * om * (1.0f + 2.0f * x) : 0.0f;
        const float l  = log1pf(dist);
        const float il = l / dist;
        s_inp[e * 4 + 0] = l;            // (dist/dist)*log1p(dist)
        s_inp[e * 4 + 1] = dx * il;
        s_inp[e * 4 + 2] = dy * il;
        s_inp[e * 4 + 3] = dz * il;
    }

    #pragma unroll 1
    for (int k = 0; k < F0; ++k) {
        const float k0 = s_kern[k], k1 = s_kern[F0 + k],
                    k2 = s_kern[2 * F0 + k], k3 = s_kern[3 * F0 + k];
        const float bk = s_bias[k];
        float ew[N_ENV];
        #pragma unroll
        for (int v = 0; v < N_ENV; ++v) ew[v] = s_envw[v * F0 + k];
        const float4 wb0 = *(const float4*)&s_wbeta[k * F1 + 0];
        const float4 wb1 = *(const float4*)&s_wbeta[k * F1 + 4];
        const float4 wb2 = *(const float4*)&s_wbeta[k * F1 + 8];
        const float4 wb3 = *(const float4*)&s_wbeta[k * F1 + 12];
        #pragma unroll
        for (int q = 0; q < 4; ++q) {
            const float a = bk + dd[q][0] * k0 + dd[q][1] * k1
                               + dd[q][2] * k2 + dd[q][3] * k3;
            // tanh(a) = 1 - 2/(exp(2a)+1); saturates correctly at +-inf
            float h = 1.0f - 2.0f / (__expf(2.0f * a) + 1.0f);
            float g = 0.0f;
            #pragma unroll
            for (int v = 0; v < N_ENV; ++v) g += evv[q][v] * ew[v];
            h *= g;
            bacc[q][0]  += h * wb0.x; bacc[q][1]  += h * wb0.y;
            bacc[q][2]  += h * wb0.z; bacc[q][3]  += h * wb0.w;
            bacc[q][4]  += h * wb1.x; bacc[q][5]  += h * wb1.y;
            bacc[q][6]  += h * wb1.z; bacc[q][7]  += h * wb1.w;
            bacc[q][8]  += h * wb2.x; bacc[q][9]  += h * wb2.y;
            bacc[q][10] += h * wb2.z; bacc[q][11] += h * wb2.w;
            bacc[q][12] += h * wb3.x; bacc[q][13] += h * wb3.y;
            bacc[q][14] += h * wb3.z; bacc[q][15] += h * wb3.w;
        }
    }

    #pragma unroll
    for (int q = 0; q < 4; ++q) {
        const int e = q * 256 + tid;
        v8h lo, hi;
        #pragma unroll
        for (int j = 0; j < 8; ++j) {
            lo[j] = (_Float16)(bacc[q][j]     * cutv[q]);
            hi[j] = (_Float16)(bacc[q][j + 8] * cutv[q]);
        }
        *(v8h*)&s_beta[e * F1]     = lo;
        *(v8h*)&s_beta[e * F1 + 8] = hi;
    }

    __syncthreads();

    // ---- Phase 2a: edge output, fully coalesced float4 stores ----
    {
        const float4* we4 = (const float4*)W_edge;                  // [c][fc] fc in 0..63
        float4* eo4 = (float4*)(edge_out + (size_t)n * N_NB * FEAT);
        #pragma unroll 1
        for (int i = tid; i < N_NB * (FEAT / 4); i += 256) {
            const int e  = i >> 6;
            const int fc = i & 63;
            const float i0 = s_inp[e * 4 + 0], i1 = s_inp[e * 4 + 1],
                        i2 = s_inp[e * 4 + 2], i3 = s_inp[e * 4 + 3];
            const float4 w0 = we4[fc], w1 = we4[64 + fc],
                         w2 = we4[128 + fc], w3 = we4[192 + fc];
            const float4 bz = *(const float4*)&s_bz[fc * 4];
            float4 o;
            o.x = bz.x + i0 * w0.x + i1 * w1.x + i2 * w2.x + i3 * w3.x;
            o.y = bz.y + i0 * w0.y + i1 * w1.y + i2 * w2.y + i3 * w3.y;
            o.z = bz.z + i0 * w0.z + i1 * w1.z + i2 * w2.z + i3 * w3.z;
            o.w = bz.w + i0 * w0.w + i1 * w1.w + i2 * w2.w + i3 * w3.w;
            eo4[(size_t)e * 64 + fc] = o;
        }
    }

    // ---- Phase 2b: gamma = beta @ W_gamma via v_wmma_f32_16x16x32_f16 ----
    // GEMM per block: M=1024 (electrons), N=256 (FEAT), K=16 (padded to 32 with zeros).
    {
        const int lane = tid & 31;
        const int wv   = tid >> 5;                                  // 8 waves
        // Preload all 16 B fragments (lane<16: column N=lane of tile, K=0..15;
        // lane>=16: K=16..31 -> zeroed pad rows of s_wgT)
        v16h Bf[16];
        #pragma unroll
        for (int nt = 0; nt < 16; ++nt) {
            const int row = (lane < 16) ? (nt * 16 + lane) : (FEAT + (lane - 16));
            Bf[nt] = *(const v16h*)&s_wgT[row * F1];
        }
        const v8h z8 = {};
        float* gbase = gamma_out + (size_t)n * N_NB * FEAT;
        #pragma unroll 1
        for (int i = 0; i < 8; ++i) {                               // 8 M-tiles per wave
            const int mt = wv * 8 + i;
            // A fragment: row M = lane&15; halfs 0..15 = K 0..15 (lane<16), zero pad K 16..31 (lane>=16)
            v16h a = {};
            if (lane < 16) a = *(const v16h*)&s_beta[(mt * 16 + (lane & 15)) * F1];
            float* gb = gbase + (size_t)(mt * 16 + (lane >> 4) * 8) * FEAT + (lane & 15);
            #pragma unroll
            for (int nt = 0; nt < 16; ++nt) {
                v8f c = {};
                v8f d = __builtin_amdgcn_wmma_f32_16x16x32_f16(
                            /*neg_a=*/false, a, /*neg_b=*/false, Bf[nt],
                            /*c_mod=*/(short)0, c,
                            /*reuse_a=*/false, /*reuse_b=*/false);
                float* gp = gb + nt * 16;
                #pragma unroll
                for (int r = 0; r < 8; ++r) gp[(size_t)r * FEAT] = d[r];
            }
        }
    }
}

extern "C" void kernel_launch(void* const* d_in, const int* in_sizes, int n_in,
                              void* d_out, int out_size, void* d_ws, size_t ws_size,
                              hipStream_t stream) {
    const float* r_nb_ne    = (const float*)d_in[0];
    const float* R          = (const float*)d_in[1];
    const float* ne_scales  = (const float*)d_in[2];
    const float* ne_kernel  = (const float*)d_in[3];
    const float* ne_bias    = (const float*)d_in[4];
    const float* env_weight = (const float*)d_in[5];
    const float* W_beta     = (const float*)d_in[6];
    const float* W_gamma    = (const float*)d_in[7];
    const float* W_edge     = (const float*)d_in[8];
    const float* b_edge     = (const float*)d_in[9];
    const float* z_n        = (const float*)d_in[10];

    float* gamma_out = (float*)d_out;
    float* edge_out  = (float*)d_out + (size_t)N_NUC * N_NB * FEAT;

    moon_gamma_edge_kernel<<<dim3(N_NUC), dim3(256), 0, stream>>>(
        r_nb_ne, R, ne_scales, ne_kernel, ne_bias, env_weight,
        W_beta, W_gamma, W_edge, b_edge, z_n, gamma_out, edge_out);
}
